// belong_weight_generator_36876589203658
// MI455X (gfx1250) — matrix operand order repaired and checked by reading, not compiled, not run on hardware
//
#include <hip/hip_runtime.h>
#include <hip/hip_bf16.h>

typedef __attribute__((ext_vector_type(16))) _Float16 v16h;
typedef __attribute__((ext_vector_type(8)))  float    v8f;

#define B_    2
#define N_    8192
#define M_    8192
#define C_    128
#define KNN   12
#define ED    60
#define CHNK  512
#define NCH   (N_ / CHNK)

#if __has_builtin(__builtin_amdgcn_global_load_async_to_lds_b64) && \
    __has_builtin(__builtin_amdgcn_s_wait_asynccnt)
#define USE_ASYNC_LDS 1
typedef int v2i_t __attribute__((vector_size(8)));
typedef __attribute__((address_space(1))) v2i_t* as1_v2i;
typedef __attribute__((address_space(3))) v2i_t* as3_v2i;
#endif

// ---------------------------------------------------------------------------
// Kernel 1: f1 = feature1 @ Wp + bp   via v_wmma_f32_16x16x32_f16
// One wave -> 16x128 output tile; block = 4 waves = 64 rows.
// B fragments preloaded 4-at-a-time to pipeline ds_load_b128 against WMMA.
// ---------------------------------------------------------------------------
__global__ __launch_bounds__(128) void k_prefix_gemm(
    const float* __restrict__ feat, const float* __restrict__ Wp,
    const float* __restrict__ bp, float* __restrict__ f1)
{
    __shared__ __align__(32) _Float16 wsw[4 * 8 * 32 * 16];   // 32 KB
    const int tid = threadIdx.x;
    for (int e = tid; e < 4 * 8 * 32 * 16; e += 128) {
        int h = e & 15, lane = (e >> 4) & 31, c = (e >> 9) & 7, s = e >> 12;
        int kg  = 32 * s + ((lane < 16) ? 0 : 16) + h;
        int col = 16 * c + (lane & 15);
        wsw[e] = (_Float16)Wp[kg * C_ + col];
    }
    __syncthreads();

    const int wv = tid >> 5, lane = tid & 31;
    const int laneLo = lane & 15, hiH = lane >> 4;
    const int base = blockIdx.x * 64 + wv * 16;
    const int rowg = base + laneLo;

    v8f acc[8];
#pragma unroll
    for (int c = 0; c < 8; ++c) {
        float bv = bp[16 * c + laneLo];
#pragma unroll
        for (int r = 0; r < 8; ++r) acc[c][r] = bv;
    }

#pragma unroll
    for (int s = 0; s < 4; ++s) {
        v16h a;
#pragma unroll
        for (int h = 0; h < 16; ++h) {
            int kg = 32 * s + h + (hiH ? 16 : 0);
            a[h] = (_Float16)feat[rowg * C_ + kg];
        }
#pragma unroll
        for (int cg = 0; cg < 2; ++cg) {
            v16h bf[4];
#pragma unroll
            for (int u = 0; u < 4; ++u)
                bf[u] = *(const v16h*)&wsw[((s * 8 + cg * 4 + u) * 32 + lane) * 16];
#pragma unroll
            for (int u = 0; u < 4; ++u)
                acc[cg * 4 + u] = __builtin_amdgcn_wmma_f32_16x16x32_f16(
                    false, a, false, bf[u], (short)0, acc[cg * 4 + u], false, false);
        }
    }

#pragma unroll
    for (int c = 0; c < 8; ++c)
#pragma unroll
        for (int r = 0; r < 8; ++r)
            f1[(base + r + 8 * hiH) * C_ + 16 * c + laneLo] = acc[c][r];
}

// ---------------------------------------------------------------------------
// Kernel 2: brute-force KNN, top-12 per query.
// Double-buffered point staging: async global->LDS copies tracked by ASYNCcnt
// (fallback: vectorized sync staging). One query per thread; register-resident
// fully-unrolled sorted insertion.
// ---------------------------------------------------------------------------
__device__ __forceinline__ void stage_chunk(const float* __restrict__ src,
                                            int ch, float* dst, int t)
{
    const float* g = src + ch * (3 * CHNK);
#ifdef USE_ASYNC_LDS
#pragma unroll
    for (int r = 0; r < 3; ++r) {
        const int i = r * 512 + t * 2;
        __builtin_amdgcn_global_load_async_to_lds_b64(
            (as1_v2i)(g + i), (as3_v2i)(dst + i), 0, 0);
    }
#else
#pragma unroll
    for (int r = 0; r < 3; ++r) {
        const int i = r * 512 + t * 2;
        *(float2*)&dst[i] = *(const float2*)&g[i];
    }
#endif
}

__global__ __launch_bounds__(256) void k_knn(
    const float* __restrict__ xyz1, const float* __restrict__ qxyz,
    int* __restrict__ outIdx)
{
    const int b = blockIdx.y;
    const int t = threadIdx.x;
    const int q = blockIdx.x * 256 + t;
    const float qx = qxyz[(b * M_ + q) * 3 + 0];
    const float qy = qxyz[(b * M_ + q) * 3 + 1];
    const float qz = qxyz[(b * M_ + q) * 3 + 2];

    __shared__ __align__(16) float sbuf[2][3 * CHNK];   // 12 KB double buffer
    float best[KNN];
    int   bidx[KNN];
#pragma unroll
    for (int j = 0; j < KNN; ++j) { best[j] = 3.4e38f; bidx[j] = 0; }

    const float* src = xyz1 + (size_t)b * N_ * 3;
    stage_chunk(src, 0, sbuf[0], t);

    for (int ch = 0; ch < NCH; ++ch) {
        if (ch + 1 < NCH) stage_chunk(src, ch + 1, sbuf[(ch + 1) & 1], t);
#ifdef USE_ASYNC_LDS
        if (ch + 1 < NCH) __builtin_amdgcn_s_wait_asynccnt(3);
        else              __builtin_amdgcn_s_wait_asynccnt(0);
#endif
        __syncthreads();

        const float* s = sbuf[ch & 1];
        const int basep = ch * CHNK;
#pragma unroll 4
        for (int i = 0; i < CHNK; ++i) {
            const float dx = s[3 * i + 0] - qx;
            const float dy = s[3 * i + 1] - qy;
            const float dz = s[3 * i + 2] - qz;
            const float d = dx * dx + dy * dy + dz * dz;
            if (d < best[KNN - 1]) {
                const int pid = basep + i;
#pragma unroll
                for (int j = KNN - 1; j >= 1; --j) {
                    if (d < best[j - 1])  { best[j] = best[j - 1]; bidx[j] = bidx[j - 1]; }
                    else if (d < best[j]) { best[j] = d;           bidx[j] = pid; }
                }
                if (d < best[0]) { best[0] = d; bidx[0] = pid; }
            }
        }
        __syncthreads();
    }
#pragma unroll
    for (int j = 0; j < KNN; ++j) outIdx[(b * M_ + q) * KNN + j] = bidx[j];
}

// ---------------------------------------------------------------------------
// Kernel 3a: positional MLP + feature gather (WMMA).
// ---------------------------------------------------------------------------
__global__ __launch_bounds__(64) void k_posmlp(
    const float* __restrict__ xyz1, const float* __restrict__ qxyz,
    const int*   __restrict__ knnIdx, const float* __restrict__ f1,
    const float* __restrict__ W1, const float* __restrict__ b1,
    const float* __restrict__ W2, const float* __restrict__ b2,
    float* __restrict__ gbuf)
{
    __shared__ __align__(32) _Float16 w1p[2 * 8 * 32 * 16];   // 16 KB
    __shared__ __align__(32) _Float16 w2p[4 * 8 * 32 * 16];   // 32 KB
    __shared__ __align__(32) _Float16 h1s[2][16 * C_];        //  8 KB

    const int tid = threadIdx.x;
    for (int e = tid; e < 2 * 8 * 32 * 16; e += 64) {
        int h = e & 15, lane = (e >> 4) & 31, c = (e >> 9) & 7, s = e >> 12;
        int kg  = 32 * s + ((lane < 16) ? 0 : 16) + h;
        int col = 16 * c + (lane & 15);
        w1p[e] = (_Float16)((kg < ED) ? W1[kg * C_ + col] : 0.0f);
    }
    for (int e = tid; e < 4 * 8 * 32 * 16; e += 64) {
        int h = e & 15, lane = (e >> 4) & 31, c = (e >> 9) & 7, s = e >> 12;
        int kg  = 32 * s + ((lane < 16) ? 0 : 16) + h;
        int col = 16 * c + (lane & 15);
        w2p[e] = (_Float16)W2[kg * C_ + col];
    }
    __syncthreads();

    const int wv = tid >> 5, lane = tid & 31;
    const int laneLo = lane & 15, hiH = lane >> 4;
    const int tile = blockIdx.x * 2 + wv;
    const int row0 = tile * 16;

    const int rowE = row0 + laneLo;
    const int bE   = rowE / (M_ * KNN);
    const int remE = rowE - bE * (M_ * KNN);
    const int mE   = remE / KNN;
    const int idxE = knnIdx[rowE];
    const float dX = xyz1[(bE * N_ + idxE) * 3 + 0] - qxyz[(bE * M_ + mE) * 3 + 0];
    const float dY = xyz1[(bE * N_ + idxE) * 3 + 1] - qxyz[(bE * M_ + mE) * 3 + 1];
    const float dZ = xyz1[(bE * N_ + idxE) * 3 + 2] - qxyz[(bE * M_ + mE) * 3 + 2];

    // -------- layer 1: emb(60->64) @ W1 + b1 --------
    v8f acc[8];
#pragma unroll
    for (int c = 0; c < 8; ++c) {
        float bv = b1[16 * c + laneLo];
#pragma unroll
        for (int r = 0; r < 8; ++r) acc[c][r] = bv;
    }
#pragma unroll
    for (int s = 0; s < 2; ++s) {
        v16h a;
#pragma unroll
        for (int h = 0; h < 16; ++h) {
            int kg = 32 * s + h + (hiH ? 16 : 0);
            float v = 0.0f;
            if (kg < ED) {
                int axis = kg / 20;
                int tt   = kg - axis * 20;
                int j    = (tt < 10) ? tt : (tt - 10);
                float dv  = (axis == 0) ? dX : ((axis == 1) ? dY : dZ);
                float om  = __expf(-0.92103404f * (float)j);   // 10000^(-j/10)
                float ang = dv * om;
                v = (tt < 10) ? __sinf(ang) : __cosf(ang);
            }
            a[h] = (_Float16)v;
        }
#pragma unroll
        for (int cg = 0; cg < 2; ++cg) {
            v16h bf[4];
#pragma unroll
            for (int u = 0; u < 4; ++u)
                bf[u] = *(const v16h*)&w1p[((s * 8 + cg * 4 + u) * 32 + lane) * 16];
#pragma unroll
            for (int u = 0; u < 4; ++u)
                acc[cg * 4 + u] = __builtin_amdgcn_wmma_f32_16x16x32_f16(
                    false, a, false, bf[u], (short)0, acc[cg * 4 + u], false, false);
        }
    }

    _Float16* h1 = h1s[wv];
#pragma unroll
    for (int c = 0; c < 8; ++c)
#pragma unroll
        for (int r = 0; r < 8; ++r)
            h1[(r + 8 * hiH) * C_ + 16 * c + laneLo] = (_Float16)fmaxf(acc[c][r], 0.0f);
    __syncthreads();

    // -------- layer 2: h1 @ W2, C preloaded with gathered f1 + b2 --------
    int gIdx[8], gB[8];
#pragma unroll
    for (int r = 0; r < 8; ++r) {
        const int row = row0 + r + 8 * hiH;
        gB[r]   = row / (M_ * KNN);
        gIdx[r] = knnIdx[row];
    }
    v8f acc2[8];
#pragma unroll
    for (int c = 0; c < 8; ++c) {
        const float bv = b2[16 * c + laneLo];
#pragma unroll
        for (int r = 0; r < 8; ++r)
            acc2[c][r] = f1[(gB[r] * N_ + gIdx[r]) * C_ + 16 * c + laneLo] + bv;
    }
#pragma unroll
    for (int s = 0; s < 4; ++s) {
        v16h a;
#pragma unroll
        for (int h = 0; h < 16; ++h) {
            int kg = 32 * s + h + (hiH ? 16 : 0);
            a[h] = h1[laneLo * C_ + kg];
        }
#pragma unroll
        for (int cg = 0; cg < 2; ++cg) {
            v16h bf[4];
#pragma unroll
            for (int u = 0; u < 4; ++u)
                bf[u] = *(const v16h*)&w2p[((s * 8 + cg * 4 + u) * 32 + lane) * 16];
#pragma unroll
            for (int u = 0; u < 4; ++u)
                acc2[cg * 4 + u] = __builtin_amdgcn_wmma_f32_16x16x32_f16(
                    false, a, false, bf[u], (short)0, acc2[cg * 4 + u], false, false);
        }
    }
#pragma unroll
    for (int c = 0; c < 8; ++c)
#pragma unroll
        for (int r = 0; r < 8; ++r)
            gbuf[(row0 + r + 8 * hiH) * C_ + 16 * c + laneLo] = acc2[c][r];
}

// ---------------------------------------------------------------------------
// Kernel 3b: weight = softmax_k( g[k] . g[0] ) * 0.4, w[0] += 0.6
// ---------------------------------------------------------------------------
__global__ __launch_bounds__(256) void k_attn(
    const float* __restrict__ gbuf, float* __restrict__ outw)
{
    const int wv = threadIdx.x >> 5, lane = threadIdx.x & 31;
    const int q  = blockIdx.x * 8 + wv;
    const int rb = q * KNN;
    const float4* g4 = (const float4*)gbuf;

    const float4 a0 = g4[rb * 32 + lane];
    float w[KNN];
#pragma unroll
    for (int k = 0; k < KNN; ++k) {
        const float4 gk = g4[(rb + k) * 32 + lane];
        float p = a0.x * gk.x + a0.y * gk.y + a0.z * gk.z + a0.w * gk.w;
#pragma unroll
        for (int off = 16; off > 0; off >>= 1) p += __shfl_xor(p, off, 32);
        w[k] = p;
    }
    float mx = w[0];
#pragma unroll
    for (int k = 1; k < KNN; ++k) mx = fmaxf(mx, w[k]);
    float sum = 0.0f;
#pragma unroll
    for (int k = 0; k < KNN; ++k) { w[k] = __expf(w[k] - mx); sum += w[k]; }
    const float inv = 0.4f / sum;
#pragma unroll
    for (int k = 0; k < KNN; ++k) w[k] *= inv;
    w[0] += 0.6f;
    if (lane == 0) {
#pragma unroll
        for (int k = 0; k < KNN; ++k) outw[rb + k] = w[k];
    }
}

// ---------------------------------------------------------------------------
extern "C" void kernel_launch(void* const* d_in, const int* in_sizes, int n_in,
                              void* d_out, int out_size, void* d_ws, size_t ws_size,
                              hipStream_t stream) {
    (void)in_sizes; (void)n_in; (void)out_size; (void)ws_size;
    const float* feat = (const float*)d_in[0];
    const float* xyz1 = (const float*)d_in[1];
    const float* qxyz = (const float*)d_in[2];
    const float* Wp   = (const float*)d_in[5];
    const float* bp   = (const float*)d_in[6];
    const float* W1   = (const float*)d_in[7];
    const float* b1   = (const float*)d_in[8];
    const float* W2   = (const float*)d_in[9];
    const float* b2   = (const float*)d_in[10];

    float* outw   = (float*)d_out;
    int*   outidx = (int*)(outw + (size_t)B_ * M_ * KNN);

    float* f1   = (float*)d_ws;                    // [B,N,128]  (8 MB)
    float* gbuf = f1 + (size_t)B_ * N_ * C_;       // [B,M,12,128] (100 MB)

    k_prefix_gemm<<<(B_ * N_) / 64, 128, 0, stream>>>(feat, Wp, bp, f1);

    dim3 g2(M_ / 256, B_);
    k_knn<<<g2, 256, 0, stream>>>(xyz1, qxyz, outidx);

    k_posmlp<<<(B_ * M_ * KNN) / 32, 64, 0, stream>>>(
        xyz1, qxyz, outidx, f1, W1, b1, W2, b2, gbuf);

    k_attn<<<(B_ * M_) / 8, 256, 0, stream>>>(gbuf, outw);
}
